// Actor_44220983279638
// MI455X (gfx1250) — matrix operand order repaired and checked by reading, not compiled, not run on hardware
//
#include <hip/hip_runtime.h>
#include <math.h>

// ---- problem constants (match reference) ----
constexpr int CB = 64;        // batch
constexpr int CD = 32;        // nodes
constexpr int CDM = 128;      // model dim
constexpr int CNH = 16;       // heads
constexpr int CDKV = 16;      // head dim
constexpr int CNL = 3;        // encoder layers
constexpr int CDI = 512;      // ffn inner
constexpr int CDU = 256;      // glimpse unit
constexpr int ROWS = CB * CD; // 2048
#define NEGV (-1e9f)

typedef __attribute__((ext_vector_type(16))) _Float16 v16h;
typedef __attribute__((ext_vector_type(2)))  _Float16 v2h;
typedef __attribute__((ext_vector_type(8)))  float    v8f;

// ======================================================================
// WMMA GEMM: C[M,N] = act(A[M,K] @ B[K,N] (+bias) (+C if ACCUM)),
// K = KTILES*32 (compile-time, fully unrolled & software-pipelined).
// Block = 32x4 (4 waves). Each wave owns a 16x32 output tile (two 16x16
// WMMA accumulators sharing one A fragment). The 32x32 B k-chunk is
// staged in double-buffered LDS **transposed** ([n][k]) so each lane's
// B fragment is 16 contiguous halfs -> 2x ds_load_b128. Staging packs
// two consecutive k per n with v_cvt_pk into one ds_store_b32. One
// barrier per chunk. A optionally column-scaled by ascale[k] (fused BN).
// ======================================================================
template<int KTILES, int ACT, bool HASBIAS, bool ACCUM, bool HASASCALE>
__global__ __launch_bounds__(128)
void wmma_gemm(const float* __restrict__ A, const float* __restrict__ Bm,
               float* __restrict__ C, const float* __restrict__ bias,
               const float* __restrict__ ascale, int M, int N)
{
    constexpr int K = KTILES * 32;
    __shared__ __align__(32) _Float16 Bs[2][32 * 32];   // [buf][n][k] transposed chunk
    const int lane = threadIdx.x;
    const int wv   = threadIdx.y;
    const int tid  = wv * 32 + lane;
    const int m0   = (blockIdx.y * 4 + wv) * 16;
    const int n0   = blockIdx.x * 32;
    const int hi   = lane >> 4;                  // lane group
    const int lo   = lane & 15;

    v8f acc0 = {}, acc1 = {};
    if (ACCUM) {
        for (int r = 0; r < 8; ++r) {
            acc0[r] = C[(size_t)(m0 + r + 8 * hi) * N + n0 + lo];
            acc1[r] = C[(size_t)(m0 + r + 8 * hi) * N + n0 + 16 + lo];
        }
    }

    const float* Arow = A + (size_t)(m0 + lo) * K;
    const int kp = (tid & 15) * 2;               // k pair: rows kp, kp+1 of chunk
    const int ng = (tid >> 4) * 4;               // 4 consecutive n

    auto stage = [&](int t, int buf) {
        const float* src0 = Bm + (size_t)(t * 32 + kp) * N + n0 + ng;       // row k
        const float* src1 = src0 + N;                                       // row k+1
        float4 r0 = *(const float4*)src0;
        float4 r1 = *(const float4*)src1;
        float a0[4] = { r0.x, r0.y, r0.z, r0.w };
        float a1[4] = { r1.x, r1.y, r1.z, r1.w };
        _Float16* bt = &Bs[buf][0];
#pragma unroll
        for (int e = 0; e < 4; ++e) {
            v2h pk = { (_Float16)a0[e], (_Float16)a1[e] };                  // (k, k+1)
            *(v2h*)&bt[(ng + e) * 32 + kp] = pk;                            // [n][k] b32
        }
    };

    stage(0, 0);
#pragma unroll
    for (int t = 0; t < KTILES; ++t) {
        __syncthreads();
        if (t + 1 < KTILES) {
            __builtin_prefetch(Arow + (t + 1) * 32, 0, 1);
            stage(t + 1, (t + 1) & 1);
        }
        // A fragment, contiguous k order matching B:
        //   halfs 0..7  = A[M, k0 + 16*hi + i], halfs 8..15 = +8
        const int k0 = t * 32;
        v16h a;
        const float* ap0 = Arow + k0 + 16 * hi;
        const float* ap1 = Arow + k0 + 8 + 16 * hi;
        if (HASASCALE) {
            for (int i = 0; i < 8; ++i) a[i]     = (_Float16)(ap0[i] * ascale[k0 + 16 * hi + i]);
            for (int i = 0; i < 8; ++i) a[8 + i] = (_Float16)(ap1[i] * ascale[k0 + 8 + 16 * hi + i]);
        } else {
            for (int i = 0; i < 8; ++i) a[i]     = (_Float16)ap0[i];
            for (int i = 0; i < 8; ++i) a[8 + i] = (_Float16)ap1[i];
        }
        // B fragments: halfs i = B[k0 + 16*hi + i, n0 (+16) + lo]
        //            = BsT[(lo (+16)) * 32 + 16*hi + i]  -> contiguous v16h
        const _Float16* bs = Bs[t & 1];
        v16h b0 = *(const v16h*)&bs[lo * 32 + 16 * hi];
        v16h b1 = *(const v16h*)&bs[(16 + lo) * 32 + 16 * hi];
        acc0 = __builtin_amdgcn_wmma_f32_16x16x32_f16(false, a, false, b0,
                                                      (short)0, acc0, false, false);
        acc1 = __builtin_amdgcn_wmma_f32_16x16x32_f16(false, a, false, b1,
                                                      (short)0, acc1, false, false);
    }

    const float bv0 = HASBIAS ? bias[n0 + lo] : 0.0f;
    const float bv1 = HASBIAS ? bias[n0 + 16 + lo] : 0.0f;
    for (int r = 0; r < 8; ++r) {
        float v0 = acc0[r] + bv0;
        float v1 = acc1[r] + bv1;
        if (ACT == 1) { v0 = v0 > 0.0f ? v0 : 0.0f; v1 = v1 > 0.0f ? v1 : 0.0f; }
        C[(size_t)(m0 + r + 8 * hi) * N + n0 + lo]      = v0;
        C[(size_t)(m0 + r + 8 * hi) * N + n0 + 16 + lo] = v1;
    }
}

// ======================================================================
// embedding: x[b,d,:] = [in0,in1,(d==0)] @ emb_W + emb_b
// ======================================================================
__global__ void embed_kernel(const float* __restrict__ in, const float* __restrict__ W,
                             const float* __restrict__ bias, float* __restrict__ x)
{
    int row = blockIdx.x, t = threadIdx.x;
    int d = row % CD;
    float i0 = in[row * 2 + 0], i1 = in[row * 2 + 1];
    float pad = (d == 0) ? 1.0f : 0.0f;
    x[(size_t)row * CDM + t] = i0 * W[0 * CDM + t] + i1 * W[1 * CDM + t]
                             + pad * W[2 * CDM + t] + bias[t];
}

// ======================================================================
// encoder self-attention: one (b,h) per 32-thread wave-block
// ======================================================================
__global__ void attn_kernel(const float* __restrict__ Q, const float* __restrict__ Kt,
                            const float* __restrict__ V, float* __restrict__ O)
{
    int bh = blockIdx.x, b = bh / CNH, h = bh % CNH;
    int i = threadIdx.x;
    __shared__ float Ks[CD][CDKV], Vs[CD][CDKV];
    size_t base = ((size_t)b * CD + i) * (CNH * CDKV) + h * CDKV;
    for (int t = 0; t < CDKV; ++t) { Ks[i][t] = Kt[base + t]; Vs[i][t] = V[base + t]; }
    __syncthreads();
    float q[CDKV];
    for (int t = 0; t < CDKV; ++t) q[t] = Q[base + t];
    float s[CD]; float mx = -1e30f;
    for (int j = 0; j < CD; ++j) {
        float d = 0.f;
        for (int t = 0; t < CDKV; ++t) d += q[t] * Ks[j][t];
        d *= 0.25f;                       // 1/sqrt(16)
        s[j] = d; mx = fmaxf(mx, d);
    }
    float sum = 0.f;
    for (int j = 0; j < CD; ++j) { s[j] = __expf(s[j] - mx); sum += s[j]; }
    float inv = 1.0f / sum;
    float o[CDKV] = {};
    for (int j = 0; j < CD; ++j) {
        float p = s[j] * inv;
        for (int t = 0; t < CDKV; ++t) o[t] += p * Vs[j][t];
    }
    for (int t = 0; t < CDKV; ++t) O[base + t] = o[t];
}

// ======================================================================
// x = LayerNorm(x + r), one 128-dim row per 128-thread block
// ======================================================================
__global__ void add_ln_kernel(float* __restrict__ x, const float* __restrict__ r,
                              const float* __restrict__ g, const float* __restrict__ bta)
{
    int row = blockIdx.x, t = threadIdx.x;
    __shared__ float red[CDM];
    float v = x[(size_t)row * CDM + t] + r[(size_t)row * CDM + t];
    red[t] = v; __syncthreads();
    for (int s = 64; s > 0; s >>= 1) { if (t < s) red[t] += red[t + s]; __syncthreads(); }
    float m = red[0] / CDM; __syncthreads();
    float dv = v - m; red[t] = dv * dv; __syncthreads();
    for (int s = 64; s > 0; s >>= 1) { if (t < s) red[t] += red[t + s]; __syncthreads(); }
    float var = red[0] / CDM;
    x[(size_t)row * CDM + t] = g[t] * dv * rsqrtf(var + 1e-6f) + bta[t];
}

// ======================================================================
// GAT: za/zb dots (z @ a[k,0], z @ a[k,1]) per (row, channel)
// ======================================================================
__global__ void gat_pre_kernel(const float* __restrict__ z3, const float* __restrict__ a,
                               float* __restrict__ za, float* __restrict__ zb)
{
    int row = blockIdx.x, k = blockIdx.y, t = threadIdx.x;
    __shared__ float r0[CDM], r1[CDM];
    float zv = z3[((size_t)k * ROWS + row) * CDM + t];
    r0[t] = zv * a[(k * 2 + 0) * CDM + t];
    r1[t] = zv * a[(k * 2 + 1) * CDM + t];
    __syncthreads();
    for (int s = 64; s > 0; s >>= 1) {
        if (t < s) { r0[t] += r0[t + s]; r1[t] += r1[t + s]; }
        __syncthreads();
    }
    if (t == 0) { za[k * ROWS + row] = r0[0]; zb[k * ROWS + row] = r1[0]; }
}

// masked-softmax attention over 3 adjacency channels; one (b,i) per block
__global__ void gat_attn_kernel(const float* __restrict__ z3, const float* __restrict__ za,
                                const float* __restrict__ zb, const int* __restrict__ adj,
                                const int* __restrict__ adj_in, const int* __restrict__ adj_out,
                                float* __restrict__ out, int relu_flag)
{
    int row = blockIdx.x, b = row / CD, i = row % CD, t = threadIdx.x;
    __shared__ float p[CD], red[CD];
    float acc = 0.0f;
    for (int k = 0; k < 3; ++k) {
        const int* A = (k == 0 ? adj : (k == 1 ? adj_in : adj_out))
                       + (size_t)b * CD * CD + (size_t)i * CD;
        float zai = za[k * ROWS + row];
        if (t < CD) {
            float e = zai + zb[k * ROWS + b * CD + t];
            e = e > 0.0f ? e : 0.2f * e;          // leaky_relu(0.2)
            p[t] = (A[t] > 0) ? e : NEGV;
            red[t] = p[t];
        }
        __syncthreads();
        for (int s = 16; s > 0; s >>= 1) { if (t < s) red[t] = fmaxf(red[t], red[t + s]); __syncthreads(); }
        float mx = red[0]; __syncthreads();
        if (t < CD) { p[t] = __expf(p[t] - mx); red[t] = p[t]; }
        __syncthreads();
        for (int s = 16; s > 0; s >>= 1) { if (t < s) red[t] += red[t + s]; __syncthreads(); }
        float inv = 1.0f / red[0];
        __syncthreads();
        float o = 0.0f;
        for (int j = 0; j < CD; ++j)
            o += p[j] * z3[((size_t)k * ROWS + b * CD + j) * CDM + t];
        acc += o * inv;
        __syncthreads();
    }
    if (relu_flag) acc = acc > 0.0f ? acc : 0.0f;
    out[(size_t)row * CDM + t] = acc;
}

// ======================================================================
// BatchNorm stats in closed form: y[b,i,j]=P[b,i]+Q[b,j]+fc_b,
// sum_{i,j}(Pp_i+Q_j)^2 = D*S2p + D*S2q + 2*S1p*S1q  (per b, per channel)
// ======================================================================
__global__ void bn_stats_kernel(const float* __restrict__ P, const float* __restrict__ Q,
                                const float* __restrict__ fcb, const float* __restrict__ bng,
                                float* __restrict__ bn_scale, float* __restrict__ bn_mean)
{
    int c = blockIdx.x, b = threadIdx.x;
    float s1p = 0, s2p = 0, s1q = 0, s2q = 0;
    for (int i = 0; i < CD; ++i) {
        float pv = P[((size_t)b * CD + i) * CDM + c] + fcb[c];
        float qv = Q[((size_t)b * CD + i) * CDM + c];
        s1p += pv; s2p += pv * pv; s1q += qv; s2q += qv * qv;
    }
    float sy  = (float)CD * (s1p + s1q);
    float sy2 = (float)CD * s2p + (float)CD * s2q + 2.0f * s1p * s1q;
    __shared__ float r1[CB], r2[CB];
    r1[b] = sy; r2[b] = sy2; __syncthreads();
    for (int s = 32; s > 0; s >>= 1) {
        if (b < s) { r1[b] += r1[b + s]; r2[b] += r2[b + s]; }
        __syncthreads();
    }
    if (b == 0) {
        float Nf = (float)(CB * CD * CD);
        float m = r1[0] / Nf;
        float var = r2[0] / Nf - m * m;
        bn_mean[c] = m;
        bn_scale[c] = bng[c] * rsqrtf(var + 1e-5f);
    }
}

// ======================================================================
// glimpse: s[row] = sum_u tanh(T[row,u]) * v[u]
// ======================================================================
__global__ void glimpse_score_kernel(const float* __restrict__ T, const float* __restrict__ vvec,
                                     float* __restrict__ sout)
{
    int row = blockIdx.x, t = threadIdx.x;
    __shared__ float red[CDU];
    red[t] = tanhf(T[(size_t)row * CDU + t]) * vvec[t];
    __syncthreads();
    for (int s = 128; s > 0; s >>= 1) { if (t < s) red[t] += red[t + s]; __syncthreads(); }
    if (t == 0) sout[row] = red[0];
}

__global__ void glimpse_pool_kernel(const float* __restrict__ s, const float* __restrict__ ne,
                                    const int* __restrict__ visited, int want,
                                    float* __restrict__ g)
{
    int b = blockIdx.x, t = threadIdx.x;
    __shared__ float p[CD], red[CD];
    if (t < CD) {
        p[t] = (visited[b * CD + t] == want) ? s[b * CD + t] : NEGV;
        red[t] = p[t];
    }
    __syncthreads();
    for (int st = 16; st > 0; st >>= 1) { if (t < st) red[t] = fmaxf(red[t], red[t + st]); __syncthreads(); }
    float mx = red[0]; __syncthreads();
    if (t < CD) { p[t] = __expf(p[t] - mx); red[t] = p[t]; }
    __syncthreads();
    for (int st = 16; st > 0; st >>= 1) { if (t < st) red[t] += red[t + st]; __syncthreads(); }
    float inv = 1.0f / red[0]; __syncthreads();
    float o = 0.0f;
    for (int d = 0; d < CD; ++d) o += p[d] * ne[((size_t)b * CD + d) * CDM + t];
    g[(size_t)b * CDM + t] = o * inv;
}

__global__ void relu_kernel(float* x, int n)
{
    int i = blockIdx.x * blockDim.x + threadIdx.x;
    if (i < n) { float v = x[i]; x[i] = v > 0.0f ? v : 0.0f; }
}

// (bn_scale*fc_b + bn_b - bn_scale*bn_mean) @ ptr_We  -> sv[128]
__global__ void shiftvec_kernel(const float* __restrict__ bscale, const float* __restrict__ bmean,
                                const float* __restrict__ fcb, const float* __restrict__ bnb,
                                const float* __restrict__ We, float* __restrict__ sv)
{
    int n = threadIdx.x;
    float acc = 0.0f;
    for (int c = 0; c < CDM; ++c) {
        float vec = bscale[c] * fcb[c] + (bnb[c] - bscale[c] * bmean[c]);
        acc += vec * We[c * CDM + n];
    }
    sv[n] = acc;
}

// ======================================================================
// pointer logits: u[b,i,j] = tanh(Pe_i + Qe_j + gW_b + sv) . ptr_v, XOR-masked
// ======================================================================
__global__ void logits_kernel(const float* __restrict__ Pe, const float* __restrict__ Qe,
                              const float* __restrict__ gW, const float* __restrict__ sv,
                              const float* __restrict__ pv, const int* __restrict__ visited,
                              float* __restrict__ logits)
{
    int row = blockIdx.x, b = row / CD, i = row % CD, t = threadIdx.x;
    __shared__ float red[CDM];
    float pe = Pe[(size_t)row * CDM + t] + gW[(size_t)b * CDM + t] + sv[t];
    float pvt = pv[t];
    int vi = visited[b * CD + i];
    for (int j = 0; j < CD; ++j) {
        int mc = vi ^ visited[b * CD + j];
        red[t] = tanhf(pe + Qe[((size_t)b * CD + j) * CDM + t]) * pvt;
        __syncthreads();
        for (int s = 64; s > 0; s >>= 1) { if (t < s) red[t] += red[t + s]; __syncthreads(); }
        if (t == 0)
            logits[(size_t)b * CD * CD + i * CD + j] = (mc == 1) ? red[0] : NEGV;
        __syncthreads();
    }
}

// argmax (first-index tiebreak) + log_softmax at argmax, per batch
__global__ void select_kernel(const float* __restrict__ logits, float* __restrict__ lp_out,
                              int* __restrict__ eidx_out, int* __restrict__ valid_out)
{
    int b = blockIdx.x, t = threadIdx.x;
    const float* L = logits + (size_t)b * (CD * CD);
    __shared__ float vmax[256]; __shared__ int vidx[256];
    float bm = -1e30f; int bi = 0;
    for (int e = t; e < CD * CD; e += 256) { float v = L[e]; if (v > bm) { bm = v; bi = e; } }
    vmax[t] = bm; vidx[t] = bi; __syncthreads();
    for (int s = 128; s > 0; s >>= 1) {
        if (t < s) {
            if (vmax[t + s] > vmax[t] ||
                (vmax[t + s] == vmax[t] && vidx[t + s] < vidx[t])) {
                vmax[t] = vmax[t + s]; vidx[t] = vidx[t + s];
            }
        }
        __syncthreads();
    }
    __shared__ float mxs; __shared__ int eis;
    if (t == 0) { mxs = vmax[0]; eis = vidx[0]; }
    __syncthreads();
    float mx = mxs;
    __shared__ float red[256];
    float se = 0.0f;
    for (int e = t; e < CD * CD; e += 256) se += __expf(L[e] - mx);
    red[t] = se; __syncthreads();
    for (int s = 128; s > 0; s >>= 1) { if (t < s) red[t] += red[t + s]; __syncthreads(); }
    if (t == 0) {
        lp_out[b] = (L[eis] - mx) - logf(red[0]);
        eidx_out[b] = eis;
        valid_out[b] = (mx > -1e8f) ? 1 : 0;   // any unmasked edge exists
    }
}

__global__ void active_kernel(const int* __restrict__ validf, int* __restrict__ activef)
{
    int t = threadIdx.x;
    __shared__ int r[CB];
    r[t] = validf[t]; __syncthreads();
    for (int s = 32; s > 0; s >>= 1) { if (t < s) r[t] |= r[t + s]; __syncthreads(); }
    if (t == 0) *activef = r[0];
}

// state update + scan outputs (xs/ys always recorded; state frozen if !active)
__global__ void update_kernel(const int* __restrict__ activef, const int* __restrict__ eidx_arr,
                              const float* __restrict__ lp_arr, int* __restrict__ visited,
                              int* __restrict__ adj, int* __restrict__ adj_in,
                              int* __restrict__ adj_out, float* __restrict__ lp_sum,
                              float* __restrict__ out_xy, int step)
{
    int b = blockIdx.x, t = threadIdx.x, i = t >> 5, j = t & 31;
    int e = eidx_arr[b], xi = e >> 5, yi = e & 31;
    bool act = (*activef) != 0;
    if (t == 0) {
        out_xy[(size_t)step * 2 * CB + 0 * CB + b] = (float)xi;
        out_xy[(size_t)step * 2 * CB + 1 * CB + b] = (float)yi;
        lp_sum[b] += act ? lp_arr[b] : 0.0f;
    }
    if (!act) return;
    int* A    = adj     + (size_t)b * CD * CD;
    int* Ain  = adj_in  + (size_t)b * CD * CD;
    int* Aout = adj_out + (size_t)b * CD * CD;
    if (t == 0) {
        A[xi * CD + yi] = 1;
        visited[b * CD + xi] = 1;
        visited[b * CD + yi] = 1;
    }
    __syncthreads();
    __shared__ int t1[CD], t2[CD];
    if (t < CD) {
        t1[t] = (t == xi) ? 0 : A[xi * CD + t];   // row xi, diag zeroed
        t2[t] = (t == yi) ? 0 : A[t * CD + yi];   // col yi, diag zeroed
    }
    __syncthreads();
    if (t1[i] && t1[j]) Ain[i * CD + j]  = 1;
    if (t2[i] && t2[j]) Aout[i * CD + j] = 1;
}

__global__ void init_kernel(int* __restrict__ visited, int* __restrict__ adj,
                            int* __restrict__ adj_in, int* __restrict__ adj_out,
                            float* __restrict__ lp_sum)
{
    int b = blockIdx.x, t = threadIdx.x, i = t >> 5, j = t & 31;
    int eye = (i == j) ? 1 : 0;
    size_t idx = (size_t)b * CD * CD + t;
    adj[idx] = eye; adj_in[idx] = eye; adj_out[idx] = eye;
    if (t < CD) visited[b * CD + t] = (t == 0) ? 1 : 0;
    if (t == 0) lp_sum[b] = 0.0f;
}

__global__ void finalize_kernel(const int* __restrict__ adj, const float* __restrict__ lp_sum,
                                float* __restrict__ out)
{
    int b = blockIdx.x, t = threadIdx.x;
    out[(size_t)b * CD * CD + t] = (float)adj[(size_t)b * CD * CD + t];
    if (t == 0) out[CB * CD * CD + b] = lp_sum[b];
}

// ======================================================================
// host-side orchestration
// ======================================================================
extern "C" void kernel_launch(void* const* d_in, const int* in_sizes, int n_in,
                              void* d_out, int out_size, void* d_ws, size_t ws_size,
                              hipStream_t stream)
{
    (void)in_sizes; (void)n_in; (void)out_size; (void)ws_size;
    const float* inputs  = (const float*)d_in[0];
    const float* emb_W   = (const float*)d_in[1];
    const float* emb_b   = (const float*)d_in[2];
    const float* enc_Wq  = (const float*)d_in[3];
    const float* enc_Wk  = (const float*)d_in[4];
    const float* enc_Wv  = (const float*)d_in[5];
    const float* enc_Wo  = (const float*)d_in[6];
    const float* ln1_g   = (const float*)d_in[7];
    const float* ln1_b   = (const float*)d_in[8];
    const float* ff_W1   = (const float*)d_in[9];
    const float* ff_b1   = (const float*)d_in[10];
    const float* ff_W2   = (const float*)d_in[11];
    const float* ff_b2   = (const float*)d_in[12];
    const float* ln2_g   = (const float*)d_in[13];
    const float* ln2_b   = (const float*)d_in[14];
    const float* gat_W   = (const float*)d_in[15];
    const float* gat_a   = (const float*)d_in[16];
    const float* fc_W    = (const float*)d_in[17];
    const float* fc_b    = (const float*)d_in[18];
    const float* bn_g    = (const float*)d_in[19];
    const float* bn_b    = (const float*)d_in[20];
    const float* ge_w1   = (const float*)d_in[21];
    const float* ge_w2   = (const float*)d_in[22];
    const float* gl_v_W  = (const float*)d_in[23];
    const float* gl_v_v  = (const float*)d_in[24];
    const float* gl_u_W  = (const float*)d_in[25];
    const float* gl_u_v  = (const float*)d_in[26];
    const float* ptr_We  = (const float*)d_in[27];
    const float* ptr_Wg  = (const float*)d_in[28];
    const float* ptr_v   = (const float*)d_in[29];
    float* out = (float*)d_out;

    // ---- workspace carve-up ----
    float* ws = (float*)d_ws;
    size_t off = 0;
    auto alloc = [&](size_t n) { float* p = ws + off; off += n; return p; };
    float* x    = alloc((size_t)ROWS * CDM);        // encoder activations -> enc
    float* qb   = alloc((size_t)ROWS * 256);
    float* kb   = alloc((size_t)ROWS * 256);
    float* vb   = alloc((size_t)ROWS * 256);
    float* ob   = alloc((size_t)ROWS * 256);
    float* tmp  = alloc((size_t)ROWS * CDI);        // ffn inner / glimpse T
    float* proj = alloc((size_t)ROWS * CDM);
    float* h    = alloc((size_t)ROWS * CDM);
    float* ne   = alloc((size_t)ROWS * CDM);
    float* z3   = alloc((size_t)3 * ROWS * CDM);
    float* za   = alloc((size_t)3 * ROWS);
    float* zbv  = alloc((size_t)3 * ROWS);
    float* Pb   = alloc((size_t)ROWS * CDM);
    float* Qb   = alloc((size_t)ROWS * CDM);
    float* Peb  = alloc((size_t)ROWS * CDM);
    float* Qeb  = alloc((size_t)ROWS * CDM);
    float* bsc  = alloc(CDM);
    float* bme  = alloc(CDM);
    float* sglv = alloc(ROWS);
    float* sglu = alloc(ROWS);
    float* gvis = alloc((size_t)CB * CDM);
    float* gunv = alloc((size_t)CB * CDM);
    float* gbuf = alloc((size_t)CB * CDM);
    float* gW   = alloc((size_t)CB * CDM);
    float* sv   = alloc(CDM);
    float* logitsb = alloc((size_t)CB * CD * CD);
    float* lp_b    = alloc(CB);
    float* lp_sum  = alloc(CB);
    int* ip = (int*)(ws + off);
    int* visited = ip; ip += CB * CD;
    int* adjB    = ip; ip += CB * CD * CD;
    int* adjIn   = ip; ip += CB * CD * CD;
    int* adjOut  = ip; ip += CB * CD * CD;
    int* eidx    = ip; ip += CB;
    int* validf  = ip; ip += CB;
    int* activef = ip; ip += 1;

    const dim3 gblk(32, 4);
    // K = 128 plain GEMM
    auto gemm128 = [&](const float* A, const float* Bm, float* C, int M, int N) {
        wmma_gemm<4, 0, false, false, false><<<dim3(N / 32, M / 64), gblk, 0, stream>>>(
            A, Bm, C, nullptr, nullptr, M, N);
    };
    auto gemm256 = [&](const float* A, const float* Bm, float* C, int M, int N) {
        wmma_gemm<8, 0, false, false, false><<<dim3(N / 32, M / 64), gblk, 0, stream>>>(
            A, Bm, C, nullptr, nullptr, M, N);
    };

    // ---- embedding ----
    embed_kernel<<<ROWS, CDM, 0, stream>>>(inputs, emb_W, emb_b, x);

    // ---- encoder (3 post-LN layers) ----
    for (int l = 0; l < CNL; ++l) {
        const float* Wq = enc_Wq + (size_t)l * CDM * 256;
        const float* Wk = enc_Wk + (size_t)l * CDM * 256;
        const float* Wv = enc_Wv + (size_t)l * CDM * 256;
        const float* Wo = enc_Wo + (size_t)l * 256 * CDM;
        gemm128(x, Wq, qb, ROWS, 256);
        gemm128(x, Wk, kb, ROWS, 256);
        gemm128(x, Wv, vb, ROWS, 256);
        attn_kernel<<<CB * CNH, 32, 0, stream>>>(qb, kb, vb, ob);
        gemm256(ob, Wo, proj, ROWS, CDM);
        add_ln_kernel<<<ROWS, CDM, 0, stream>>>(x, proj, ln1_g + l * CDM, ln1_b + l * CDM);
        wmma_gemm<4, 1, true, false, false><<<dim3(CDI / 32, ROWS / 64), gblk, 0, stream>>>(
            x, ff_W1 + (size_t)l * CDM * CDI, tmp, ff_b1 + l * CDI, nullptr, ROWS, CDI);
        wmma_gemm<16, 0, true, false, false><<<dim3(CDM / 32, ROWS / 64), gblk, 0, stream>>>(
            tmp, ff_W2 + (size_t)l * CDI * CDM, proj, ff_b2 + l * CDM, nullptr, ROWS, CDM);
        add_ln_kernel<<<ROWS, CDM, 0, stream>>>(x, proj, ln2_g + l * CDM, ln2_b + l * CDM);
    }
    // x now holds enc (read-only below)

    init_kernel<<<CB, CD * CD, 0, stream>>>(visited, adjB, adjIn, adjOut, lp_sum);
    float* out_xy = out + CB * CD * CD + CB;

    // ---- 31 scan steps ----
    for (int s = 0; s < CD - 1; ++s) {
        // GAT layer 1 (input enc, relu)
        for (int k = 0; k < 3; ++k)
            gemm128(x, gat_W + (size_t)(0 * 3 + k) * CDM * CDM, z3 + (size_t)k * ROWS * CDM,
                    ROWS, CDM);
        gat_pre_kernel<<<dim3(ROWS, 3), CDM, 0, stream>>>(z3, gat_a + 0 * 3 * 2 * CDM, za, zbv);
        gat_attn_kernel<<<ROWS, CDM, 0, stream>>>(z3, za, zbv, adjB, adjIn, adjOut, h, 1);
        // GAT layer 2 (input h, no relu)
        for (int k = 0; k < 3; ++k)
            gemm128(h, gat_W + (size_t)(1 * 3 + k) * CDM * CDM, z3 + (size_t)k * ROWS * CDM,
                    ROWS, CDM);
        gat_pre_kernel<<<dim3(ROWS, 3), CDM, 0, stream>>>(z3, gat_a + 1 * 3 * 2 * CDM, za, zbv);
        gat_attn_kernel<<<ROWS, CDM, 0, stream>>>(z3, za, zbv, adjB, adjIn, adjOut, ne, 0);

        // rank-structured edge embedding: P = ne@fc_W[:128], Q = ne@fc_W[128:]
        gemm128(ne, fc_W, Pb, ROWS, CDM);
        gemm128(ne, fc_W + (size_t)CDM * CDM, Qb, ROWS, CDM);
        bn_stats_kernel<<<CDM, CB, 0, stream>>>(Pb, Qb, fc_b, bn_g, bsc, bme);

        // glimpses
        gemm128(ne, gl_v_W, tmp, ROWS, CDU);
        glimpse_score_kernel<<<ROWS, CDU, 0, stream>>>(tmp, gl_v_v, sglv);
        glimpse_pool_kernel<<<CB, CDM, 0, stream>>>(sglv, ne, visited, 1, gvis);
        gemm128(ne, gl_u_W, tmp, ROWS, CDU);
        glimpse_score_kernel<<<ROWS, CDU, 0, stream>>>(tmp, gl_u_v, sglu);
        glimpse_pool_kernel<<<CB, CDM, 0, stream>>>(sglu, ne, visited, 0, gunv);

        // g = relu(g_unv@ge_w1 + g_vis@ge_w2)
        gemm128(gunv, ge_w1, gbuf, CB, CDM);
        wmma_gemm<4, 0, false, true, false><<<dim3(CDM / 32, CB / 64), gblk, 0, stream>>>(
            gvis, ge_w2, gbuf, nullptr, nullptr, CB, CDM);
        relu_kernel<<<(CB * CDM + 255) / 256, 256, 0, stream>>>(gbuf, CB * CDM);
        gemm128(gbuf, ptr_Wg, gW, CB, CDM);

        // pointer pre-activations, BN folded: Pe=(s∘P)@We, Qe=(s∘Q)@We, sv=const@We
        shiftvec_kernel<<<1, CDM, 0, stream>>>(bsc, bme, fc_b, bn_b, ptr_We, sv);
        wmma_gemm<4, 0, false, false, true><<<dim3(CDM / 32, ROWS / 64), gblk, 0, stream>>>(
            Pb, ptr_We, Peb, nullptr, bsc, ROWS, CDM);
        wmma_gemm<4, 0, false, false, true><<<dim3(CDM / 32, ROWS / 64), gblk, 0, stream>>>(
            Qb, ptr_We, Qeb, nullptr, bsc, ROWS, CDM);

        logits_kernel<<<ROWS, CDM, 0, stream>>>(Peb, Qeb, gW, sv, ptr_v, visited, logitsb);
        select_kernel<<<CB, 256, 0, stream>>>(logitsb, lp_b, eidx, validf);
        active_kernel<<<1, CB, 0, stream>>>(validf, activef);
        update_kernel<<<CB, CD * CD, 0, stream>>>(activef, eidx, lp_b, visited, adjB, adjIn,
                                                  adjOut, lp_sum, out_xy, s);
    }

    finalize_kernel<<<CB, CD * CD, 0, stream>>>(adjB, lp_sum, out);
}
